// OuterProduct_11441792877252
// MI455X (gfx1250) — matrix operand order repaired and checked by reading, not compiled, not run on hardware
//
#include <hip/hip_runtime.h>
#include <hip/hip_bf16.h>

typedef __attribute__((ext_vector_type(16))) _Float16 v16h;
typedef __attribute__((ext_vector_type(8)))  _Float16 v8h;
typedef __attribute__((ext_vector_type(8)))  float    v8f;

#define N_SEQ 768
#define C_S   384
#define C_H   64
#define C_Z   128

// ---------------------------------------------------------------------------
// Kernel 1: convert wo [c_z, c_h] fp32 -> f16 (B-operand source for WMMA)
// ---------------------------------------------------------------------------
__global__ void cvt_wo_kernel(const float* __restrict__ wo, _Float16* __restrict__ wf) {
    int idx = blockIdx.x * blockDim.x + threadIdx.x;
    if (idx < C_Z * C_H) wf[idx] = (_Float16)wo[idx];
}

// ---------------------------------------------------------------------------
// Kernel 2: per-row LayerNorm + two [384]->[64] projections.
// One block per sequence row; 128 threads. Row cached in LDS, weights stream
// from L2 (w1+w2 = 196 KB, resident in the 192 MB L2 after first touch).
// Outputs a,b stored as f16 for the WMMA stage.
// ---------------------------------------------------------------------------
__global__ void ln_ab_kernel(const float* __restrict__ s,
                             const float* __restrict__ ln_w,
                             const float* __restrict__ ln_b,
                             const float* __restrict__ w1,
                             const float* __restrict__ b1,
                             const float* __restrict__ w2,
                             const float* __restrict__ b2,
                             _Float16* __restrict__ af,
                             _Float16* __restrict__ bf) {
    __shared__ float lnrow[C_S];
    __shared__ float red1[128];
    __shared__ float red2[128];

    const int t   = threadIdx.x;
    const int row = blockIdx.x;
    const float* srow = s + (size_t)row * C_S;

    float sum = 0.f, sumsq = 0.f;
    for (int c = t; c < C_S; c += 128) {
        float v = srow[c];
        lnrow[c] = v;
        sum += v;
        sumsq += v * v;
    }
    red1[t] = sum;
    red2[t] = sumsq;
    __syncthreads();
    #pragma unroll
    for (int off = 64; off > 0; off >>= 1) {
        if (t < off) { red1[t] += red1[t + off]; red2[t] += red2[t + off]; }
        __syncthreads();
    }
    const float mean = red1[0] * (1.0f / C_S);
    const float var  = red2[0] * (1.0f / C_S) - mean * mean;
    const float rstd = rsqrtf(var + 1e-5f);
    __syncthreads();

    for (int c = t; c < C_S; c += 128)
        lnrow[c] = (lnrow[c] - mean) * rstd * ln_w[c] + ln_b[c];
    __syncthreads();

    // threads 0..63 -> a[h=t], threads 64..127 -> b[h=t-64]
    const int h = t & 63;
    const float* wrow = (t < 64 ? w1 : w2) + (size_t)h * C_S;
    float acc = (t < 64 ? b1 : b2)[h];
    const float4* w4 = (const float4*)wrow;
    const float4* l4 = (const float4*)lnrow;
    #pragma unroll 8
    for (int c = 0; c < C_S / 4; ++c) {
        float4 wv = w4[c];
        float4 lv = l4[c];
        acc += wv.x * lv.x + wv.y * lv.y + wv.z * lv.z + wv.w * lv.w;
    }
    (t < 64 ? af : bf)[(size_t)row * C_H + h] = (_Float16)acc;
}

// ---------------------------------------------------------------------------
// Kernel 3: out[i,j,z] = sum_h a[i,h]*b[j,h]*wo[z,h] + bo[z]
// GEMM view: M = (i,j) pairs, K = 64, N = 128, via v_wmma_f32_16x16x32_f16.
// Workgroup = 16x16 (i,j) tile x full z (8 waves).
// Wave w owns z-tile nt = w (keeps only 2 W fragments live -> no spills)
// and loops over the 16 i-rows of the tile.
// A-fragment for an M-tile (i fixed, j = j0..j0+15):
//   A[m,k] = a[i,k] * b[j0+m,k]  (packed f16 muls, built in fragment layout).
// ISA fragment layouts (wave32):
//   A 16x32 f16: lane m=l&15; e<8 -> k=e+8*khi, e>=8 -> k=e+8+8*khi (khi=l>>4)
//   B 32x16 f16: lane n=l&15; k same as A (e<8 -> k=e+8*khi, e>=8 -> k=e+8+8*khi)
//   C/D 16x16 f32: lane n=l&15; vgpr r -> m = r + 8*khi
// ---------------------------------------------------------------------------
__device__ __forceinline__ v16h cat8(v8h lo, v8h hi) {
    return __builtin_shufflevector(lo, hi, 0, 1, 2, 3, 4, 5, 6, 7,
                                            8, 9, 10, 11, 12, 13, 14, 15);
}

__global__ void outer_wmma_kernel(const _Float16* __restrict__ af,
                                  const _Float16* __restrict__ bf,
                                  const _Float16* __restrict__ wf,
                                  const float* __restrict__ bo,
                                  float* __restrict__ out) {
    const int lane = threadIdx.x & 31;
    const int wave = threadIdx.x >> 5;   // = z-tile index nt (0..7)
    const int khi  = (lane >> 4) & 1;    // lane half
    const int m    = lane & 15;          // row/col within fragment
    const int i0   = blockIdx.y * 16;
    const int j0   = blockIdx.x * 16;

    // b-row fragments (loop-invariant): row j0+m, k chunks 0..31 / 32..63
    const _Float16* brow = bf + (size_t)(j0 + m) * C_H;
    const v16h Bv0 = cat8(*(const v8h*)(brow +      8 * khi),
                          *(const v8h*)(brow + 16 + 8 * khi));
    const v16h Bv1 = cat8(*(const v8h*)(brow + 32 + 8 * khi),
                          *(const v8h*)(brow + 48 + 8 * khi));

    // W fragments for this wave's z-tile: rows z = wave*16 + n (n = m)
    const _Float16* wrow = wf + (size_t)(wave * 16 + m) * C_H;
    const v16h W0 = cat8(*(const v8h*)(wrow +      8 * khi),
                         *(const v8h*)(wrow + 16 + 8 * khi));
    const v16h W1 = cat8(*(const v8h*)(wrow + 32 + 8 * khi),
                         *(const v8h*)(wrow + 48 + 8 * khi));

    const float bz = bo[wave * 16 + m];

    #pragma unroll 2
    for (int mt = 0; mt < 16; ++mt) {
        const int i = i0 + mt;
        const _Float16* arow = af + (size_t)i * C_H;
        const v16h Av0 = cat8(*(const v8h*)(arow +      8 * khi),
                              *(const v8h*)(arow + 16 + 8 * khi));
        const v16h Av1 = cat8(*(const v8h*)(arow + 32 + 8 * khi),
                              *(const v8h*)(arow + 48 + 8 * khi));
        const v16h A0 = Av0 * Bv0;   // A fragment, k 0..31
        const v16h A1 = Av1 * Bv1;   // A fragment, k 32..63

        v8f c = {};
        c = __builtin_amdgcn_wmma_f32_16x16x32_f16(false, A0, false, W0,
                                                   (short)0, c, false, false);
        c = __builtin_amdgcn_wmma_f32_16x16x32_f16(false, A1, false, W1,
                                                   (short)0, c, false, false);

        float* obase = out + ((size_t)i * N_SEQ + j0 + 8 * khi) * C_Z
                           + wave * 16 + m;
        #pragma unroll
        for (int r = 0; r < 8; ++r) {
            // row m_out = r + 8*khi -> j = j0 + r + 8*khi; col z = wave*16 + m
            __builtin_nontemporal_store(c[r] + bz, obase + (size_t)r * C_Z);
        }
    }
}

// ---------------------------------------------------------------------------
extern "C" void kernel_launch(void* const* d_in, const int* in_sizes, int n_in,
                              void* d_out, int out_size, void* d_ws, size_t ws_size,
                              hipStream_t stream) {
    const float* s    = (const float*)d_in[0];
    // d_in[1] = timestep (unused on the layernorm path)
    const float* ln_w = (const float*)d_in[2];
    const float* ln_b = (const float*)d_in[3];
    const float* w1   = (const float*)d_in[4];
    const float* b1   = (const float*)d_in[5];
    const float* w2   = (const float*)d_in[6];
    const float* b2   = (const float*)d_in[7];
    const float* wo   = (const float*)d_in[8];
    const float* bo   = (const float*)d_in[9];
    float* out = (float*)d_out;

    _Float16* af = (_Float16*)d_ws;              // [768][64] f16
    _Float16* bf = af + (size_t)N_SEQ * C_H;     // [768][64] f16
    _Float16* wf = bf + (size_t)N_SEQ * C_H;     // [128][64] f16

    cvt_wo_kernel<<<(C_Z * C_H + 255) / 256, 256, 0, stream>>>(wo, wf);
    ln_ab_kernel<<<N_SEQ, 128, 0, stream>>>(s, ln_w, ln_b, w1, b1, w2, b2, af, bf);
    outer_wmma_kernel<<<dim3(N_SEQ / 16, N_SEQ / 16), 256, 0, stream>>>(af, bf, wf, bo, out);
}
